// OptimizedLinear_72026601554094
// MI455X (gfx1250) — matrix operand order repaired and checked by reading, not compiled, not run on hardware
//
#include <hip/hip_runtime.h>

typedef __attribute__((ext_vector_type(4)))  float    v4f;
typedef __attribute__((ext_vector_type(8)))  float    v8f;
typedef __attribute__((ext_vector_type(4)))  _Float16 v4h;
typedef __attribute__((ext_vector_type(8)))  _Float16 v8h;
typedef __attribute__((ext_vector_type(16))) _Float16 v16h;

// Problem shape (from reference setup_inputs)
constexpr int BATCH = 16;
constexpr int SEQ   = 2048;
constexpr int KIN   = 1024;
constexpr int NOUT  = 4096;

// Block tile
constexpr int BM = 128;
constexpr int BN = 128;
constexpr int BK = 32;            // one f16-WMMA K-step per chunk
constexpr int LDH = BK + 8;       // padded LDS stride in halves: 40 -> 80B rows, 16B-aligned, conflict-free
constexpr int NCHUNK = KIN / BK;  // 32

__global__ __launch_bounds__(256)
void fused_sim_linear(const float* __restrict__ x,
                      const float* __restrict__ W,
                      const float* __restrict__ bias,
                      const float* __restrict__ Gw,
                      const float* __restrict__ Gb,
                      const float* __restrict__ lrs,
                      float* __restrict__ out)
{
    // Double-buffered split-fp16 tiles (hi/lo of x, hi/lo of W - lr*Gw)
    __shared__ _Float16 sAh[2][BM * LDH];
    __shared__ _Float16 sAl[2][BM * LDH];
    __shared__ _Float16 sBh[2][BN * LDH];
    __shared__ _Float16 sBl[2][BN * LDH];

    const int bb = blockIdx.z;
    const int m0 = blockIdx.y * BM;
    const int n0 = blockIdx.x * BN;
    const float lr = lrs[bb];            // uniform -> scalar load

    const int tid   = threadIdx.x;
    const int lane  = tid & 31;
    const int wave  = tid >> 5;
    const int waveM = (wave & 1) * 64;   // 2 waves along M (4 m-tiles each)
    const int waveN = (wave >> 1) * 32;  // 4 waves along N (2 n-tiles each)

    const int lm = lane & 15;            // row within a 16x16 tile
    const int lh = lane >> 4;            // half-wave selector for 16-bit K grouping

    v8f acc[4][2];
    #pragma unroll
    for (int i = 0; i < 4; ++i)
        #pragma unroll
        for (int j = 0; j < 2; ++j)
            #pragma unroll
            for (int e = 0; e < 8; ++e)
                acc[i][j][e] = 0.0f;

    // Cooperative fill: 32 floats/row = 8 x float4; 8 threads/row, 4 passes x 32 rows.
    const int rowF = tid >> 3;
    const int colF = (tid & 7) * 4;

    const float* xb = x + (size_t)bb * SEQ * KIN;

    // Prefetch registers for next K-chunk
    v4f ra[4], rw[4], rg[4];

    auto load_chunk = [&](int k0) {
        #pragma unroll
        for (int p = 0; p < 4; ++p) {
            const int r = rowF + p * 32;
            ra[p] = *(const v4f*)(xb + (size_t)(m0 + r) * KIN + k0 + colF);
            rw[p] = *(const v4f*)(W  + (size_t)(n0 + r) * KIN + k0 + colF);
            rg[p] = *(const v4f*)(Gw + (size_t)(n0 + r) * KIN + k0 + colF);
        }
    };

    auto split_store = [&](_Float16* dh, _Float16* dl, v4f v) {
        v4h hi, lo;
        #pragma unroll
        for (int e = 0; e < 4; ++e) {
            hi[e] = (_Float16)v[e];                   // RNE: top ~11 mantissa bits
            lo[e] = (_Float16)(v[e] - (float)hi[e]);  // next ~11 bits
        }
        *(v4h*)dh = hi;
        *(v4h*)dl = lo;
    };

    auto store_chunk = [&](int buf) {
        #pragma unroll
        for (int p = 0; p < 4; ++p) {
            const int r = rowF + p * 32;
            split_store(&sAh[buf][r * LDH + colF], &sAl[buf][r * LDH + colF], ra[p]);
            v4f c = rw[p] - rg[p] * lr;               // simulated weight, fused
            split_store(&sBh[buf][r * LDH + colF], &sBl[buf][r * LDH + colF], c);
        }
    };

    // Fragment fetch per ISA 16-bit layouts.
    // A 16x32: lanes<16 take K {0..7,16..23}, lanes>=16 take K {8..15,24..31}.
    auto fragA = [&](const _Float16* base) -> v16h {
        v8h p0 = *(const v8h*)(base + 8 * lh);
        v8h p1 = *(const v8h*)(base + 16 + 8 * lh);
        return __builtin_shufflevector(p0, p1, 0,1,2,3,4,5,6,7,8,9,10,11,12,13,14,15);
    };
    // B 32x16: lanes<16 take K {0..7,16..23}, lanes>=16 take K {8..15,24..31} (column = lane&15).
    auto fragB = [&](const _Float16* base) -> v16h {
        v8h p0 = *(const v8h*)(base + 8 * lh);
        v8h p1 = *(const v8h*)(base + 16 + 8 * lh);
        return __builtin_shufflevector(p0, p1, 0,1,2,3,4,5,6,7,8,9,10,11,12,13,14,15);
    };

    auto compute_chunk = [&](int buf) {
        v16h Ah[4], Al[4], Bh[2], Bl[2];
        #pragma unroll
        for (int mt = 0; mt < 4; ++mt) {
            const int row = (waveM + mt * 16 + lm) * LDH;
            Ah[mt] = fragA(&sAh[buf][row]);
            Al[mt] = fragA(&sAl[buf][row]);
        }
        #pragma unroll
        for (int nt = 0; nt < 2; ++nt) {
            const int row = (waveN + nt * 16 + lm) * LDH;
            Bh[nt] = fragB(&sBh[buf][row]);
            Bl[nt] = fragB(&sBl[buf][row]);
        }
        // Three split products, interleaved across 8 independent accumulators.
        #pragma unroll
        for (int mt = 0; mt < 4; ++mt)
            #pragma unroll
            for (int nt = 0; nt < 2; ++nt)
                acc[mt][nt] = __builtin_amdgcn_wmma_f32_16x16x32_f16(
                    false, Ah[mt], false, Bh[nt], (short)0, acc[mt][nt], false, false);
        #pragma unroll
        for (int mt = 0; mt < 4; ++mt)
            #pragma unroll
            for (int nt = 0; nt < 2; ++nt)
                acc[mt][nt] = __builtin_amdgcn_wmma_f32_16x16x32_f16(
                    false, Ah[mt], false, Bl[nt], (short)0, acc[mt][nt], false, false);
        #pragma unroll
        for (int mt = 0; mt < 4; ++mt)
            #pragma unroll
            for (int nt = 0; nt < 2; ++nt)
                acc[mt][nt] = __builtin_amdgcn_wmma_f32_16x16x32_f16(
                    false, Al[mt], false, Bh[nt], (short)0, acc[mt][nt], false, false);
    };

    // Pipeline prologue: fill buffer 0
    load_chunk(0);
    store_chunk(0);
    __syncthreads();

    int cur = 0;
    #pragma unroll 1
    for (int i = 0; i < NCHUNK - 1; ++i) {
        load_chunk((i + 1) * BK);   // global loads in flight during compute
        compute_chunk(cur);
        store_chunk(cur ^ 1);       // waits loads, fills the other buffer
        __syncthreads();            // single barrier per iteration
        cur ^= 1;
    }
    compute_chunk(cur);

    // Epilogue: simulated bias + store. C/D layout: VGPR r holds M=r (lanes 0-15)
    // or M=r+8 (lanes 16-31); N = tile_n*16 + (lane&15).
    float* ob = out + (size_t)bb * SEQ * NOUT;
    const int mhalf = lh * 8;
    #pragma unroll
    for (int nt = 0; nt < 2; ++nt) {
        const int gN = n0 + waveN + nt * 16 + lm;
        const float sb = bias[gN] - lr * Gb[gN];
        #pragma unroll
        for (int mt = 0; mt < 4; ++mt) {
            #pragma unroll
            for (int r = 0; r < 8; ++r) {
                const int gM = m0 + waveM + mt * 16 + mhalf + r;
                ob[(size_t)gM * NOUT + gN] = acc[mt][nt][r] + sb;
            }
        }
    }
}

extern "C" void kernel_launch(void* const* d_in, const int* in_sizes, int n_in,
                              void* d_out, int out_size, void* d_ws, size_t ws_size,
                              hipStream_t stream) {
    (void)in_sizes; (void)n_in; (void)out_size; (void)d_ws; (void)ws_size;
    const float* x    = (const float*)d_in[0];
    const float* W    = (const float*)d_in[1];
    const float* bias = (const float*)d_in[2];
    const float* Gw   = (const float*)d_in[3];
    const float* Gb   = (const float*)d_in[4];
    const float* lrs  = (const float*)d_in[5];
    float* out = (float*)d_out;

    dim3 grid(NOUT / BN, SEQ / BM, BATCH);   // (32, 16, 16)
    dim3 block(256);
    fused_sim_linear<<<grid, block, 0, stream>>>(x, W, bias, Gw, Gb, lrs, out);
}
